// CNN_49967649521704
// MI455X (gfx1250) — matrix operand order repaired and checked by reading, not compiled, not run on hardware
//
#include <hip/hip_runtime.h>
#include <hip/hip_bf16.h>

typedef _Float16 v16h __attribute__((ext_vector_type(16)));
typedef _Float16 v8h  __attribute__((ext_vector_type(8)));
typedef float    v8f  __attribute__((ext_vector_type(8)));

#define BB   512
#define SS   128
#define SPAD 134          // 3 halo rows + 128 + 3 halo rows
#define CINP 1024         // CIN=1000 padded to 1024
#define DC   256
#define NTAP 15           // 3 + 5 + 7
#define VR   19
#define FTOT 2904         // 300 + 300 + 2304

// ---------------------------------------------------------------------------
// Kernel 1: assemble padded f16 input  X[b, row=0..133, ci=0..1023]
//   row r -> sequence position s = r-3 ; rows outside [0,128) are zero halo.
//   ci: [0,900) = W, [900,950) = pos_emb1[idx1], [950,1000) = pos_emb2[idx2]
// ---------------------------------------------------------------------------
__global__ void prep_x_kernel(const float* __restrict__ W,
                              const int* __restrict__ p1,
                              const int* __restrict__ p2,
                              const float* __restrict__ pe1,
                              const float* __restrict__ pe2,
                              _Float16* __restrict__ X) {
    int row = blockIdx.x % SPAD;
    int b   = blockIdx.x / SPAD;
    _Float16* dst = X + ((size_t)b * SPAD + row) * CINP;
    int s = row - 3;
    if (s < 0 || s >= SS) {
        for (int i = threadIdx.x; i < CINP; i += 256) dst[i] = (_Float16)0.0f;
        return;
    }
    int i1 = p1[b * SS + s];
    int i2 = p2[b * SS + s];
    const float* wrow = W + ((size_t)b * SS + s) * 900;
    for (int ci = threadIdx.x; ci < CINP; ci += 256) {
        float v;
        if (ci < 900)       v = wrow[ci];
        else if (ci < 950)  v = pe1[i1 * 50 + (ci - 900)];
        else if (ci < 1000) v = pe2[i2 * 50 + (ci - 950)];
        else                v = 0.0f;
        dst[ci] = (_Float16)v;
    }
}

// ---------------------------------------------------------------------------
// Kernel 2: weights [DC, 1000, k] (k=3,5,7) -> f16 [tap(0..14)][co][ci(1024)]
// ---------------------------------------------------------------------------
__global__ void prep_w_kernel(const float* __restrict__ w3,
                              const float* __restrict__ w5,
                              const float* __restrict__ w7,
                              _Float16* __restrict__ Wf) {
    int tap = blockIdx.x >> 8;     // 0..14
    int co  = blockIdx.x & 255;
    const float* src; int k, j;
    if (tap < 3)      { src = w3; k = 3; j = tap; }
    else if (tap < 8) { src = w5; k = 5; j = tap - 3; }
    else              { src = w7; k = 7; j = tap - 8; }
    _Float16* dst = Wf + ((size_t)tap * DC + co) * CINP;
    for (int ci = threadIdx.x; ci < CINP; ci += 256) {
        float v = (ci < 1000) ? src[((size_t)co * 1000 + ci) * k + j] : 0.0f;
        dst[ci] = (_Float16)v;
    }
}

// ---------------------------------------------------------------------------
// Kernel 3: conv (implicit GEMM via WMMA) + bias + tanh + piecewise pooling.
// One workgroup (16 waves, 512 threads) per (b, kernel-size).
// X[b] staged fully in LDS (274 KB of the 320 KB/WGP).
// wave = (nhalf<<3)|mwave : owns M-tiles {mwave, mwave+8} (B-frag reuse x2)
// and N-tiles nhalf*4 .. nhalf*4+3.  Accumulators: 2x4 tiles = 64 VGPRs/lane,
// leaving headroom for the scheduler to multi-buffer B fragments (fewer
// WMMA->VALU WAR hazard NOPs) at 4 waves/SIMD occupancy.
// ---------------------------------------------------------------------------
__global__ void __launch_bounds__(512, 1)
conv_pool_kernel(const _Float16* __restrict__ X,
                 const _Float16* __restrict__ Wf,
                 const float* __restrict__ cb3,
                 const float* __restrict__ cb5,
                 const float* __restrict__ cb7,
                 const int* __restrict__ e1p,
                 const int* __restrict__ e2p,
                 float* __restrict__ pooled) {
    extern __shared__ char smem_raw[];
    _Float16* Xs  = (_Float16*)smem_raw;                       // 134*1024 f16 = 274432 B
    float*    pool = (float*)(smem_raw + SPAD * CINP * 2);     // 768 f32

    int b   = blockIdx.x / 3;
    int kid = blockIdx.x % 3;
    int tid   = threadIdx.x;
    int lane  = tid & 31;
    int wave  = tid >> 5;        // 0..15
    int mwave = wave & 7;        // M-tile group
    int nhalf = wave >> 3;       // N-tile half
    int lrow  = lane & 15;
    int lhalf = lane >> 4;

    // ---- stage X[b] into LDS (16B vector copies) ----
    {
        const uint4* src = (const uint4*)(X + (size_t)b * SPAD * CINP);
        uint4* dst = (uint4*)Xs;
        const int n16 = SPAD * CINP / 8;  // 17152
        for (int i = tid; i < n16; i += 512) dst[i] = src[i];
    }
    for (int i = tid; i < DC * 3; i += 512) pool[i] = 0.0f;
    __syncthreads();

    const int kk  = (kid == 0) ? 3 : (kid == 1) ? 5 : 7;
    const int t0  = (kid == 0) ? 0 : (kid == 1) ? 3 : 8;
    const int pad = kk >> 1;
    const float* cb = (kid == 0) ? cb3 : (kid == 1) ? cb5 : cb7;

    v8f acc0[4], acc1[4];
#pragma unroll
    for (int nt = 0; nt < 4; ++nt) { acc0[nt] = 0.0f; acc1[nt] = 0.0f; }

    const int nbase = nhalf * 4;     // first N-tile owned by this wave

    for (int j = 0; j < kk; ++j) {
        // A fragments: row = co, per-lane 16 contiguous K at lhalf*16
        const _Float16* a0 = Wf + ((size_t)((t0 + j) * DC + mwave * 16 + lrow)) * CINP + lhalf * 16;
        const _Float16* a1 = Wf + ((size_t)((t0 + j) * DC + (mwave + 8) * 16 + lrow)) * CINP + lhalf * 16;
        const int rshift = j - pad + 3;   // halo-adjusted row shift, always in range
        for (int kc = 0; kc < 32; ++kc) {
            const int cio = kc * 32;
            v8h a0lo = *(const v8h*)(a0 + cio);
            v8h a0hi = *(const v8h*)(a0 + cio + 8);
            v8h a1lo = *(const v8h*)(a1 + cio);
            v8h a1hi = *(const v8h*)(a1 + cio + 8);
            v16h A0 = __builtin_shufflevector(a0lo, a0hi, 0,1,2,3,4,5,6,7,8,9,10,11,12,13,14,15);
            v16h A1 = __builtin_shufflevector(a1lo, a1hi, 0,1,2,3,4,5,6,7,8,9,10,11,12,13,14,15);
            const int cib = cio + lhalf * 16;
#pragma unroll
            for (int nt = 0; nt < 4; ++nt) {
                // B fragment: column N = l = (nbase+nt)*16 + lrow, 16 contiguous ci
                const _Float16* bp = Xs + (size_t)((nbase + nt) * 16 + lrow + rshift) * CINP + cib;
                v16h Bf = *(const v16h*)bp;
                acc0[nt] = __builtin_amdgcn_wmma_f32_16x16x32_f16(
                    false, A0, false, Bf, (short)0, acc0[nt], false, false);
                acc1[nt] = __builtin_amdgcn_wmma_f32_16x16x32_f16(
                    false, A1, false, Bf, (short)0, acc1[nt], false, false);
            }
        }
    }

    // ---- bias + tanh + segment accumulation (LDS atomics, wg-private) ----
    int pa = e1p[b], pb2 = e2p[b];
    int d1 = min(pa, pb2), d2 = max(pa, pb2);
#pragma unroll
    for (int mt = 0; mt < 2; ++mt) {
        int cobase = ((mt ? (mwave + 8) : mwave) * 16) + (lhalf << 3);
#pragma unroll
        for (int nt = 0; nt < 4; ++nt) {
            v8f a = mt ? acc1[nt] : acc0[nt];
            int l = (nbase + nt) * 16 + lrow;
            int seg = (l < d1) ? 0 : (l < d2) ? 1 : (l < SS - 1) ? 2 : -1;
#pragma unroll
            for (int r = 0; r < 8; ++r) {
                int co = cobase + r;
                float v = tanhf(a[r] + cb[co]);
                if (seg >= 0) atomicAdd(&pool[co * 3 + seg], v);
            }
        }
    }
    __syncthreads();

    float c1 = (float)max(d1, 1);
    float c2 = (float)max(d2 - d1, 1);
    float c3 = (float)max(SS - 1 - d2, 1);
    if (tid < DC) {
        float* outp = pooled + (size_t)b * (3 * DC * 3) + kid * (DC * 3) + tid * 3;
        outp[0] = pool[tid * 3 + 0] / c1;
        outp[1] = pool[tid * 3 + 1] / c2;
        outp[2] = pool[tid * 3 + 2] / c3;
    }
}

// ---------------------------------------------------------------------------
// Kernel 4: FC (2904 -> 19) + softmax.  One block per batch, wave per row.
// ---------------------------------------------------------------------------
__global__ void fc_softmax_kernel(const float* __restrict__ e1,
                                  const float* __restrict__ e2,
                                  const float* __restrict__ pooled,
                                  const float* __restrict__ fcw,
                                  const float* __restrict__ fcb,
                                  float* __restrict__ out) {
    int b = blockIdx.x;
    __shared__ float logits[VR];
    int lane = threadIdx.x & 31;
    int wave = threadIdx.x >> 5;
    for (int v = wave; v < VR; v += 8) {
        float acc = 0.0f;
        for (int f = lane; f < FTOT; f += 32) {
            float x;
            if (f < 300)      x = e1[b * 300 + f];
            else if (f < 600) x = e2[b * 300 + (f - 300)];
            else              x = pooled[(size_t)b * 2304 + (f - 600)];
            acc += x * fcw[v * FTOT + f];
        }
        for (int off = 16; off > 0; off >>= 1) acc += __shfl_xor(acc, off, 32);
        if (lane == 0) logits[v] = acc + fcb[v];
    }
    __syncthreads();
    if (threadIdx.x == 0) {
        float m = -3.0e38f;
        for (int v = 0; v < VR; ++v) m = fmaxf(m, logits[v]);
        float e[VR]; float sum = 0.0f;
        for (int v = 0; v < VR; ++v) { e[v] = __expf(logits[v] - m); sum += e[v]; }
        float inv = 1.0f / sum;
        for (int v = 0; v < VR; ++v) out[b * VR + v] = e[v] * inv;
    }
}

// ---------------------------------------------------------------------------
extern "C" void kernel_launch(void* const* d_in, const int* in_sizes, int n_in,
                              void* d_out, int out_size, void* d_ws, size_t ws_size,
                              hipStream_t stream) {
    const float* W    = (const float*)d_in[0];
    const int*   Wp1  = (const int*)  d_in[1];
    const int*   Wp2  = (const int*)  d_in[2];
    const float* e1   = (const float*)d_in[3];
    const float* e2   = (const float*)d_in[4];
    const int*   e1p  = (const int*)  d_in[5];
    const int*   e2p  = (const int*)  d_in[6];
    const float* pe1  = (const float*)d_in[7];
    const float* pe2  = (const float*)d_in[8];
    const float* fcw  = (const float*)d_in[9];
    const float* fcb  = (const float*)d_in[10];
    const float* cw3  = (const float*)d_in[11];
    const float* cb3  = (const float*)d_in[12];
    const float* cw5  = (const float*)d_in[13];
    const float* cb5  = (const float*)d_in[14];
    const float* cw7  = (const float*)d_in[15];
    const float* cb7  = (const float*)d_in[16];

    char* ws = (char*)d_ws;
    const size_t xBytes = (size_t)BB * SPAD * CINP * 2;          // 140,509,184
    const size_t wBytes = (size_t)NTAP * DC * CINP * 2;          //   7,864,320
    _Float16* Xf     = (_Float16*)ws;
    _Float16* Wf     = (_Float16*)(ws + xBytes);
    float*    pooled = (float*)  (ws + xBytes + wBytes);         // B*2304 f32

    prep_x_kernel<<<BB * SPAD, 256, 0, stream>>>(W, Wp1, Wp2, pe1, pe2, Xf);
    prep_w_kernel<<<NTAP * DC, 256, 0, stream>>>(cw3, cw5, cw7, Wf);

    const size_t shmem = (size_t)SPAD * CINP * 2 + DC * 3 * sizeof(float); // 277,504 B
    (void)hipFuncSetAttribute((const void*)conv_pool_kernel,
                              hipFuncAttributeMaxDynamicSharedMemorySize, (int)shmem);
    conv_pool_kernel<<<BB * 3, 512, shmem, stream>>>(Xf, Wf, cb3, cb5, cb7,
                                                     e1p, e2p, pooled);

    fc_softmax_kernel<<<BB, 256, 0, stream>>>(e1, e2, pooled, fcw, fcb, (float*)d_out);
}
